// RingLlamaAttention_19481971655176
// MI455X (gfx1250) — matrix operand order repaired and checked by reading, not compiled, not run on hardware
//
#include <hip/hip_runtime.h>
#include <hip/hip_bf16.h>
#include <stdint.h>

#define HIDDEN 4096
#define SEQ    2048
#define NHEADS 32
#define NKV    8
#define HDIM   128

typedef __attribute__((ext_vector_type(16))) __bf16 v16bf;
typedef __attribute__((ext_vector_type(8)))  float  v8f;

union AFrag { v16bf v; unsigned int u[8]; };
union CAcc  { v8f  v; float f[8]; };

__device__ inline unsigned short f2bfu(float x){
  union { __bf16 b; unsigned short u; } t; t.b = (__bf16)x; return t.u;
}
__device__ inline unsigned int pack2bf(float a, float b){
  return (unsigned int)f2bfu(a) | ((unsigned int)f2bfu(b) << 16);
}

// ---- async global->LDS copy of 16 bytes (falls back to sync copy) ----
typedef int v4i_gcc __attribute__((vector_size(16)));
typedef __attribute__((address_space(1))) v4i_gcc* gptr_v4i;
typedef __attribute__((address_space(3))) v4i_gcc* lptr_v4i;

__device__ inline void async_copy16(const __bf16* g, __bf16* l){
#if defined(__gfx1250__) && __has_builtin(__builtin_amdgcn_global_load_async_to_lds_b128)
  __builtin_amdgcn_global_load_async_to_lds_b128(
      (gptr_v4i)(uintptr_t)g,
      (lptr_v4i)(unsigned int)(uintptr_t)l,
      0, 0);
#else
  *(uint4*)l = *(const uint4*)g;
#endif
}
__device__ inline void wait_async(){
#if defined(__gfx1250__) && __has_builtin(__builtin_amdgcn_s_wait_asynccnt)
  __builtin_amdgcn_s_wait_asynccnt(0);
#elif defined(__gfx1250__)
  asm volatile("s_wait_asynccnt 0" ::: "memory");
#endif
}

// =====================================================================
// GEMM: C[M,N] = A[M,K] * W[N,K]^T
// Block: 256x128, BK=32, double-buffered LDS, 8 waves x (32x128) tiles
// (2 A-frags + 8 B-frags -> 16 WMMAs per K-step per wave).
// A fp32 (hidden states, converted in-flight) or bf16 (async-to-LDS).
// Optional RoPE epilogue (wave owns a full head-aligned 128 columns).
// =====================================================================
template<bool ABF16, bool OUTBF16, bool ROPE>
__global__ __launch_bounds__(256) void gemm_bf16_wmma(
    const float* __restrict__ Af, const __bf16* __restrict__ Ab,
    const float* __restrict__ W,
    float* __restrict__ Of, __bf16* __restrict__ Ob,
    const int* __restrict__ pos, int M, int N, int K)
{
  constexpr int LDA = 40;                         // 32 + 8 pad (bf16 elems)
  __shared__ __align__(16) __bf16 As[2][256*LDA]; // 2 x 20.0 KB
  __shared__ __align__(16) __bf16 Ws[2][128*LDA]; // 2 x 10.0 KB
  const int tid = threadIdx.x;
  const int wv  = tid >> 5, ln = tid & 31;
  const int m0  = blockIdx.y * 256;
  const int n0  = blockIdx.x * 128;
  const int lcol  = ln & 15;
  const int half8 = (ln >= 16) ? 8 : 0;

  CAcc acc[2][8];
  #pragma unroll
  for (int fi = 0; fi < 2; ++fi)
    #pragma unroll
    for (int t = 0; t < 8; ++t)
      #pragma unroll
      for (int j = 0; j < 8; ++j) acc[fi][t].f[j] = 0.f;

  float4 ga[8];   // fp32 A staging (256x32 / 256 thr = 8 float4)
  float4 gw[4];   // fp32 W staging (128x32 / 256 thr = 4 float4)

  auto loadA = [&](int kk){
    #pragma unroll
    for (int i = 0; i < 8; ++i) {
      int v = tid + i*256; int r = v >> 3; int c = (v & 7) * 4;
      ga[i] = *(const float4*)&Af[(size_t)(m0 + r)*K + kk + c];
    }
  };
  auto storeA = [&](int b){
    #pragma unroll
    for (int i = 0; i < 8; ++i) {
      int v = tid + i*256; int r = v >> 3; int c = (v & 7) * 4;
      unsigned int* p = (unsigned int*)&As[b][r*LDA + c];
      p[0] = pack2bf(ga[i].x, ga[i].y); p[1] = pack2bf(ga[i].z, ga[i].w);
    }
  };
  auto asyncA = [&](int kk, int b){     // bf16 A: pure byte copy, async path
    #pragma unroll
    for (int i = 0; i < 4; ++i) {
      int v = tid + i*256; int r = v >> 2; int c = (v & 3) * 8;
      async_copy16(&Ab[(size_t)(m0 + r)*K + kk + c], &As[b][r*LDA + c]);
    }
  };
  auto loadW = [&](int kk){
    #pragma unroll
    for (int i = 0; i < 4; ++i) {
      int v = tid + i*256; int r = v >> 3; int c = (v & 7) * 4;
      gw[i] = *(const float4*)&W[(size_t)(n0 + r)*K + kk + c];
    }
  };
  auto storeW = [&](int b){
    #pragma unroll
    for (int i = 0; i < 4; ++i) {
      int v = tid + i*256; int r = v >> 3; int c = (v & 7) * 4;
      unsigned int* p = (unsigned int*)&Ws[b][r*LDA + c];
      p[0] = pack2bf(gw[i].x, gw[i].y); p[1] = pack2bf(gw[i].z, gw[i].w);
    }
  };

  // ---- pipeline prologue: stage K-step 0 into buffer 0 ----
  if (ABF16) asyncA(0, 0); else { loadA(0); storeA(0); }
  loadW(0); storeW(0);
  wait_async();
  __syncthreads();

  int buf = 0;
  for (int kk = 0; kk < K; kk += 32) {
    const int kn = kk + 32;
    const bool has_next = kn < K;
    // issue next tile's global traffic under this tile's WMMAs
    if (has_next) {
      if (ABF16) asyncA(kn, buf ^ 1); else loadA(kn);
      loadW(kn);
    }
    // ---- compute from current buffer ----
    AFrag a0, a1;
    {
      int ko = (ln < 16) ? 0 : 8;
      const __bf16* b0 = &As[buf][(wv*32 + lcol)*LDA];
      const __bf16* b1 = &As[buf][(wv*32 + 16 + lcol)*LDA];
      #pragma unroll
      for (int j = 0; j < 4; ++j) {
        a0.u[j]     = *(const unsigned int*)&b0[ko + 2*j];
        a0.u[4 + j] = *(const unsigned int*)&b0[ko + 16 + 2*j];
        a1.u[j]     = *(const unsigned int*)&b1[ko + 2*j];
        a1.u[4 + j] = *(const unsigned int*)&b1[ko + 16 + 2*j];
      }
    }
    #pragma unroll
    for (int t = 0; t < 8; ++t) {
      AFrag b;
      int ko = (ln < 16) ? 0 : 8;
      const __bf16* base = &Ws[buf][(t*16 + lcol)*LDA + ko];
      #pragma unroll
      for (int j = 0; j < 8; ++j) b.u[j] = *(const unsigned int*)&base[(j < 4) ? 2*j : 16 + 2*(j - 4)];
      acc[0][t].v = __builtin_amdgcn_wmma_f32_16x16x32_bf16(
          false, a0.v, false, b.v, (short)0, acc[0][t].v, false, false);
      acc[1][t].v = __builtin_amdgcn_wmma_f32_16x16x32_bf16(
          false, a1.v, false, b.v, (short)0, acc[1][t].v, false, false);
    }
    // ---- publish next buffer ----
    if (has_next) {
      if (!ABF16) storeA(buf ^ 1);
      storeW(buf ^ 1);
      wait_async();
    }
    __syncthreads();
    buf ^= 1;
  }

  // ---- RoPE epilogue (per-wave 128 cols == one head) ----
  if (ROPE) {
    const float KR = 0.20503700f;   // 2*ln(500000)/128
    #pragma unroll
    for (int t = 0; t < 4; ++t) {
      int d = t*16 + lcol;
      float invf = __expf(-(float)d * KR);
      #pragma unroll
      for (int fi = 0; fi < 2; ++fi)
        #pragma unroll
        for (int r = 0; r < 8; ++r) {
          int m = m0 + wv*32 + fi*16 + r + half8;
          float p = (float)pos[m];
          float c, s; __sincosf(p * invf, &c, &s);
          float x = acc[fi][t].f[r], y = acc[fi][t + 4].f[r];
          acc[fi][t].f[r]     = x*c - y*s;
          acc[fi][t + 4].f[r] = y*c + x*s;
        }
    }
  }
  // ---- store ----
  #pragma unroll
  for (int fi = 0; fi < 2; ++fi)
    #pragma unroll
    for (int t = 0; t < 8; ++t)
      #pragma unroll
      for (int r = 0; r < 8; ++r) {
        int m = m0 + wv*32 + fi*16 + r + half8;
        int n = n0 + t*16 + lcol;
        if (OUTBF16) Ob[(size_t)m*N + n] = (__bf16)acc[fi][t].f[r];
        else         Of[(size_t)m*N + n] = acc[fi][t].f[r];
      }
}

// =====================================================================
// Flash attention, dense (ring-LSE recombination == global softmax).
// Block = 128 q rows x 1 head; 8 waves x 16 rows; 32-key tiles,
// double-buffered K (async-to-LDS) and V (register-staged transpose).
// =====================================================================
__global__ __launch_bounds__(256) void attn_fa_wmma(
    const __bf16* __restrict__ Q, const __bf16* __restrict__ Kc,
    const __bf16* __restrict__ Vc, __bf16* __restrict__ O)
{
  constexpr int LDK = 136;   // K tile [32 keys][128 dims] + pad
  constexpr int LDV = 40;    // V tile transposed [128 dims][32 keys] + pad
  constexpr int LDP = 40;    // P staging [16 rows][32 keys] + pad
  __shared__ __align__(16) __bf16 Kt[2][32*LDK];
  __shared__ __align__(16) __bf16 Vt[2][128*LDV];
  __shared__ __align__(16) __bf16 Ps[8*16*LDP];

  const int tid = threadIdx.x, wv = tid >> 5, ln = tid & 31;
  const int head = blockIdx.y, kvh = head >> 2;
  const int q0   = blockIdx.x * 128 + wv * 16;
  const int lcol = ln & 15;
  const int half8 = (ln >= 16) ? 8 : 0;
  const float SCALE = 0.08838834764831845f;    // 1/sqrt(128)

  // ---- Q fragments: 16 rows x 128 dims as 4 A-fragments ----
  AFrag qf[4];
  {
    int m = q0 + lcol;
    const __bf16* qrow = &Q[(size_t)m*HIDDEN + head*HDIM];
    int ko = (ln < 16) ? 0 : 8;
    #pragma unroll
    for (int i = 0; i < 4; ++i)
      #pragma unroll
      for (int j = 0; j < 4; ++j) {
        qf[i].u[j]     = *(const unsigned int*)&qrow[i*32 + ko + 2*j];
        qf[i].u[4 + j] = *(const unsigned int*)&qrow[i*32 + ko + 16 + 2*j];
      }
  }

  CAcc acc[8];
  float rmax[8], rsum[8];
  #pragma unroll
  for (int t = 0; t < 8; ++t)
    #pragma unroll
    for (int j = 0; j < 8; ++j) acc[t].f[j] = 0.f;
  #pragma unroll
  for (int r = 0; r < 8; ++r) { rmax[r] = -1e30f; rsum[r] = 0.f; }

  uint4 gv[2];   // V staging: 16 bf16 per thread
  auto asyncK = [&](int kb, int b){
    int key = tid >> 3, d0 = (tid & 7) * 16;
    const __bf16* src = &Kc[(size_t)(kb*32 + key)*(NKV*HDIM) + kvh*HDIM + d0];
    __bf16* dst = &Kt[b][key*LDK + d0];
    async_copy16(src,     dst);
    async_copy16(src + 8, dst + 8);
  };
  auto loadV = [&](int kb){
    int key = tid >> 3, d0 = (tid & 7) * 16;
    const uint4* src = (const uint4*)&Vc[(size_t)(kb*32 + key)*(NKV*HDIM) + kvh*HDIM + d0];
    gv[0] = src[0]; gv[1] = src[1];
  };
  auto storeV = [&](int b){
    int key = tid >> 3, d0 = (tid & 7) * 16;
    const __bf16* s = (const __bf16*)gv;
    #pragma unroll
    for (int j = 0; j < 16; ++j) Vt[b][(d0 + j)*LDV + key] = s[j];
  };

  // ---- pipeline prologue ----
  asyncK(0, 0);
  loadV(0); storeV(0);
  wait_async();
  __syncthreads();

  int buf = 0;
  for (int kb = 0; kb < SEQ/32; ++kb) {
    const bool has_next = (kb + 1) < SEQ/32;
    if (has_next) { asyncK(kb + 1, buf ^ 1); loadV(kb + 1); }

    // ---- scores: 2 tiles of 16x16 keys, K-dim 128 = 4 WMMAs each ----
    CAcc s0, s1;
    #pragma unroll
    for (int j = 0; j < 8; ++j) { s0.f[j] = 0.f; s1.f[j] = 0.f; }
    #pragma unroll
    for (int i = 0; i < 4; ++i) {
      AFrag b0, b1;
      int ko = (ln < 16) ? 0 : 8;
      {
        const __bf16* base = &Kt[buf][lcol*LDK + i*32 + ko];
        #pragma unroll
        for (int j = 0; j < 8; ++j) b0.u[j] = *(const unsigned int*)&base[(j < 4) ? 2*j : 16 + 2*(j - 4)];
      }
      {
        const __bf16* base = &Kt[buf][(16 + lcol)*LDK + i*32 + ko];
        #pragma unroll
        for (int j = 0; j < 8; ++j) b1.u[j] = *(const unsigned int*)&base[(j < 4) ? 2*j : 16 + 2*(j - 4)];
      }
      s0.v = __builtin_amdgcn_wmma_f32_16x16x32_bf16(false, qf[i].v, false, b0.v, (short)0, s0.v, false, false);
      s1.v = __builtin_amdgcn_wmma_f32_16x16x32_bf16(false, qf[i].v, false, b1.v, (short)0, s1.v, false, false);
    }

    // ---- online softmax (row reductions inside each 16-lane half) ----
    #pragma unroll
    for (int r = 0; r < 8; ++r) {
      float a0 = s0.f[r]*SCALE, a1 = s1.f[r]*SCALE;
      float tm = fmaxf(a0, a1);
      #pragma unroll
      for (int msk = 8; msk >= 1; msk >>= 1) tm = fmaxf(tm, __shfl_xor(tm, msk, 16));
      float nm = fmaxf(rmax[r], tm);
      float alpha = __expf(rmax[r] - nm);
      rmax[r] = nm;
      float p0 = __expf(a0 - nm), p1 = __expf(a1 - nm);
      float ps = p0 + p1;
      #pragma unroll
      for (int msk = 8; msk >= 1; msk >>= 1) ps += __shfl_xor(ps, msk, 16);
      rsum[r] = rsum[r]*alpha + ps;
      #pragma unroll
      for (int t = 0; t < 8; ++t) acc[t].f[r] *= alpha;
      // P -> LDS (row = r + half8; keys across lanes)
      __bf16* prow = &Ps[(wv*16 + r + half8)*LDP];
      prow[lcol]      = (__bf16)p0;
      prow[16 + lcol] = (__bf16)p1;
    }

    // ---- P fragment (A layout; same wave's LDS strip, DS in-order) ----
    AFrag pf;
    {
      int ko = (ln < 16) ? 0 : 8;
      const __bf16* base = &Ps[(wv*16 + lcol)*LDP];
      #pragma unroll
      for (int j = 0; j < 4; ++j) {
        pf.u[j]     = *(const unsigned int*)&base[ko + 2*j];
        pf.u[4 + j] = *(const unsigned int*)&base[ko + 16 + 2*j];
      }
    }
    // ---- PV: 8 dim tiles ----
    #pragma unroll
    for (int t = 0; t < 8; ++t) {
      AFrag vb;
      int dcol = t*16 + lcol;
      int ko = (ln < 16) ? 0 : 8;
      const __bf16* base = &Vt[buf][dcol*LDV + ko];
      #pragma unroll
      for (int j = 0; j < 8; ++j) vb.u[j] = *(const unsigned int*)&base[(j < 4) ? 2*j : 16 + 2*(j - 4)];
      acc[t].v = __builtin_amdgcn_wmma_f32_16x16x32_bf16(false, pf.v, false, vb.v, (short)0, acc[t].v, false, false);
    }

    if (has_next) { storeV(buf ^ 1); wait_async(); }
    __syncthreads();
    buf ^= 1;
  }

  // ---- normalize + store bf16 ----
  #pragma unroll
  for (int t = 0; t < 8; ++t)
    #pragma unroll
    for (int r = 0; r < 8; ++r) {
      int m = q0 + r + half8;
      float o = acc[t].f[r] / rsum[r];
      O[(size_t)m*HIDDEN + head*HDIM + t*16 + lcol] = (__bf16)o;
    }
}

// =====================================================================
extern "C" void kernel_launch(void* const* d_in, const int* in_sizes, int n_in,
                              void* d_out, int out_size, void* d_ws, size_t ws_size,
                              hipStream_t stream)
{
  (void)in_sizes; (void)n_in; (void)out_size; (void)ws_size;
  const float* hs = (const float*)d_in[0];
  const int*   pos = (const int*)d_in[1];
  const float* wq = (const float*)d_in[2];
  const float* wk = (const float*)d_in[3];
  const float* wvp = (const float*)d_in[4];
  const float* wo = (const float*)d_in[5];
  float* out = (float*)d_out;

  __bf16* qb   = (__bf16*)d_ws;                        // 2048*4096
  __bf16* kbuf = qb   + (size_t)SEQ*HIDDEN;            // 2048*1024
  __bf16* vbuf = kbuf + (size_t)SEQ*NKV*HDIM;          // 2048*1024
  __bf16* abuf = vbuf + (size_t)SEQ*NKV*HDIM;          // 2048*4096

  dim3 blk(256);
  // Q = hs @ Wq^T  (+RoPE)
  gemm_bf16_wmma<false, true, true><<<dim3(HIDDEN/128, SEQ/256), blk, 0, stream>>>(
      hs, nullptr, wq, nullptr, qb, pos, SEQ, HIDDEN, HIDDEN);
  // K = hs @ Wk^T  (+RoPE)
  gemm_bf16_wmma<false, true, true><<<dim3((NKV*HDIM)/128, SEQ/256), blk, 0, stream>>>(
      hs, nullptr, wk, nullptr, kbuf, pos, SEQ, NKV*HDIM, HIDDEN);
  // V = hs @ Wv^T
  gemm_bf16_wmma<false, true, false><<<dim3((NKV*HDIM)/128, SEQ/256), blk, 0, stream>>>(
      hs, nullptr, wvp, nullptr, vbuf, pos, SEQ, NKV*HDIM, HIDDEN);
  // attention (dense softmax == ring recombination)
  attn_fa_wmma<<<dim3(SEQ/128, NHEADS), blk, 0, stream>>>(qb, kbuf, vbuf, abuf);
  // out = attn @ Wo^T  (fp32 out, bf16 A staged via async-to-LDS)
  gemm_bf16_wmma<true, false, false><<<dim3(HIDDEN/128, SEQ/256), blk, 0, stream>>>(
      nullptr, abuf, wo, out, nullptr, nullptr, SEQ, HIDDEN, HIDDEN);
}
